// TemporalAttentionLayer_1614907703990
// MI455X (gfx1250) — matrix operand order repaired and checked by reading, not compiled, not run on hardware
//
#include <hip/hip_runtime.h>
#include <cstdint>
#include <cstddef>

// ---------------------------------------------------------------------------
// TemporalAttentionLayer on gfx1250 (MI455X), bf16 WMMA pipeline + TDM.
// B=16384, N=3, IN=1024, OUT=256, H=8.  Rows R = B*N = 49152.
// ---------------------------------------------------------------------------

#define B_SZ     16384
#define N_SEQ    3
#define IN_DIM   1024
#define OUT_DIM  256
#define HEADS    8
#define ROWS     (B_SZ * N_SEQ)          // 49152
#define FEAT     (HEADS * OUT_DIM)       // 2048
#define QKV_N    (3 * HEADS * OUT_DIM)   // 6144
#define NEGINF   (-1e9f)

typedef __attribute__((ext_vector_type(16))) __bf16 v16bf;
typedef __attribute__((ext_vector_type(8)))  __bf16 v8bf;
typedef __attribute__((ext_vector_type(8)))  float  v8f;
typedef __attribute__((ext_vector_type(4)))  unsigned int u32x4;
typedef __attribute__((ext_vector_type(8)))  int          i32x8;
typedef __attribute__((ext_vector_type(4)))  int          i32x4;

__device__ __forceinline__ uint16_t f2bf(float f) {
  union { float f; uint32_t u; } v; v.f = f;
  uint32_t u = v.u;
  u += 0x7FFFu + ((u >> 16) & 1u);      // round-to-nearest-even
  return (uint16_t)(u >> 16);
}

__device__ __forceinline__ v16bf cat16(v8bf lo, v8bf hi) {
  return __builtin_shufflevector(lo, hi, 0,1,2,3,4,5,6,7,8,9,10,11,12,13,14,15);
}

// A-fragment (16x32 bf16, row-major src with row stride `ld` elements).
// ISA layout: lanes 0-15 row M=lane, K runs [0,8)+[16,24); lanes 16-31 same M,
// K runs [8,16)+[24,32).  -> two contiguous 16B loads per lane.
__device__ __forceinline__ v16bf load_a_frag(const uint16_t* base, int ld,
                                             int lr, int lh, int k0) {
  const uint16_t* p = base + (size_t)lr * ld + k0 + 8 * lh;
  v8bf lo = *reinterpret_cast<const v8bf*>(p);
  v8bf hi = *reinterpret_cast<const v8bf*>(p + 16);
  return cat16(lo, hi);
}

// B-fragment (32x16 bf16) from W^T stored [N][K] row-major, row stride `ld`.
// ISA layout: lanes 0-15 col N=lane, K [0,8)+[16,24); lanes 16-31 K [8,16)+[24,32).
__device__ __forceinline__ v16bf load_b_frag(const uint16_t* base, int ld,
                                             int lr, int lh, int k0) {
  const uint16_t* p = base + (size_t)lr * ld + k0 + 8 * lh;
  v8bf lo = *reinterpret_cast<const v8bf*>(p);
  v8bf hi = *reinterpret_cast<const v8bf*>(p + 16);
  return cat16(lo, hi);
}

#define WMMA_BF16(a, b, c) \
  __builtin_amdgcn_wmma_f32_16x16x32_bf16(false, (a), false, (b), (short)0, (c), false, false)

// ---------------------------------------------------------------------------
// Prep 1: xp = x + pos_emb, f32 -> bf16, row-major [ROWS][IN_DIM].
// ---------------------------------------------------------------------------
__global__ void k_prep_xp(const float* __restrict__ x,
                          const float* __restrict__ pos,
                          uint16_t* __restrict__ xp) {
  size_t e = ((size_t)blockIdx.x * blockDim.x + threadIdx.x) * 8;
  if (e >= (size_t)ROWS * IN_DIM) return;
  size_t r = e >> 10;                 // / IN_DIM
  int    k = (int)(e & (IN_DIM - 1));
  int    n = (int)(r % N_SEQ);
  const float4* xv = reinterpret_cast<const float4*>(x + e);
  const float4* pv = reinterpret_cast<const float4*>(pos + (size_t)n * IN_DIM + k);
  float4 a0 = xv[0], p0 = pv[0], a1 = xv[1], p1 = pv[1];
  uint4 o;
  o.x = (uint32_t)f2bf(a0.x + p0.x) | ((uint32_t)f2bf(a0.y + p0.y) << 16);
  o.y = (uint32_t)f2bf(a0.z + p0.z) | ((uint32_t)f2bf(a0.w + p0.w) << 16);
  o.z = (uint32_t)f2bf(a1.x + p1.x) | ((uint32_t)f2bf(a1.y + p1.y) << 16);
  o.w = (uint32_t)f2bf(a1.z + p1.z) | ((uint32_t)f2bf(a1.w + p1.w) << 16);
  *reinterpret_cast<uint4*>(xp + e) = o;
}

// ---------------------------------------------------------------------------
// Prep 2: pack Wq/Wk/Wv [H][K][O] f32 -> WqkvT [(p*8+h)*256+o][K] bf16.
// ---------------------------------------------------------------------------
__global__ void k_pack_wqkv(const float* __restrict__ Wq,
                            const float* __restrict__ Wk,
                            const float* __restrict__ Wv,
                            uint16_t* __restrict__ wt) {
  size_t idx = (size_t)blockIdx.x * blockDim.x + threadIdx.x;
  if (idx >= (size_t)QKV_N * IN_DIM) return;
  int c = (int)(idx >> 10);           // column (0..6143)
  int k = (int)(idx & (IN_DIM - 1));
  int p = c >> 11;                    // 0=q,1=k,2=v
  int h = (c >> 8) & 7;
  int o = c & 255;
  const float* W = (p == 0) ? Wq : (p == 1) ? Wk : Wv;
  wt[idx] = f2bf(W[((size_t)h * IN_DIM + k) * OUT_DIM + o]);
}

// ---------------------------------------------------------------------------
// Prep 3: pack Wf [FEAT][OUT] f32 -> WfT [OUT][FEAT] bf16.
// ---------------------------------------------------------------------------
__global__ void k_pack_wf(const float* __restrict__ Wf, uint16_t* __restrict__ wt) {
  size_t idx = (size_t)blockIdx.x * blockDim.x + threadIdx.x;
  if (idx >= (size_t)OUT_DIM * FEAT) return;
  int o = (int)(idx >> 11);           // / FEAT
  int f = (int)(idx & (FEAT - 1));
  wt[idx] = f2bf(Wf[(size_t)f * OUT_DIM + o]);
}

// ---------------------------------------------------------------------------
// Fused QKV GEMM + masked softmax-over-i attention + head-major feats write.
// grid = (B/32, HEADS); block = 512 (16 waves).  Workgroup: 32 batches
// (96 rows = 6 M-tiles) x one head (256 cols); wave w owns cols [w*16,+16).
// A (xp) is staged global->LDS by the Tensor Data Mover (double-buffered,
// TDM pad feature gives 80B LDS row stride = bank-conflict-free ds_loads).
// acc: 3 qkv x 6 mt = 18 tiles (144 VGPRs); A transient from LDS.
// ---------------------------------------------------------------------------
#define BT        32                  // batches per workgroup
#define MROWS     (BT * N_SEQ)        // 96
#define MT        6                   // M-tiles of 16
#define LDSTR     260                 // padded LDS row stride for q/k/v (f32)
#define LDS_QKV   (MROWS * LDSTR)     // per-matrix floats (24960)
#define SC_F      (3 * LDS_QKV)       // float index of scores
#define ABUF_F    (SC_F + BT * 9)     // float index of A staging buffers
#define ABUF_BYTE (ABUF_F * 4)        // byte offset of A buffers in LDS
#define A_ROWE    40                  // LDS A row stride in elements (32+8 pad)
#define ABUF_E    (MROWS * A_ROWE)    // elements per A buffer (3840)
#define ABUF_B    (ABUF_E * 2)        // bytes per A buffer (7680)
#define LDS_BYTES (ABUF_BYTE + 2 * ABUF_B)   // 316,032 B of 320 KB

// One TDM descriptor-driven DMA: 96x32 bf16 tile (row stride 1024 elements)
// from global `gsrc` into LDS byte offset `lds_off`, with 4-DW pad inserted
// after every 16 DW (-> 80B LDS row pitch).  D# layout per CDNA5 ISA 8.3/8.4.
// clang-23 toolchain: 6-arg builtin (g0, g1, g2, g3, g4, cpol).
__device__ __forceinline__ void tdm_load_a(const uint16_t* gsrc, unsigned lds_off) {
  unsigned long long ga = (unsigned long long)(uintptr_t)gsrc;
  u32x4 g0;
  g0[0] = 1u;                                          // count=1, user mode
  g0[1] = lds_off;                                     // lds_addr
  g0[2] = (unsigned)(ga & 0xFFFFFFFFu);                // global_addr[31:0]
  g0[3] = (unsigned)((ga >> 32) & 0x01FFFFFFu)         // global_addr[56:32]
          | (2u << 30);                                // type=2 ("image")
  i32x8 g1;
  g1[0] = (int)((1u << 16)                             // data_size = 2 bytes
              | (1u << 20)                             // pad_enable
              | (3u << 22)                             // pad_interval = 16 DW
              | (3u << 25));                           // pad_amount  = 4 DW
  g1[1] = (int)(32u << 16);                            // tensor_dim0 = 32
  g1[2] = (int)((unsigned)MROWS << 16);                // tensor_dim1 = 96
  g1[3] = (int)(32u << 16);                            // tile_dim0 = 32
  g1[4] = (int)MROWS;                                  // tile_dim1 = 96
  g1[5] = (int)IN_DIM;                                 // tensor_dim0_stride
  g1[6] = 0;
  g1[7] = 0;
  i32x4 z4 = {0, 0, 0, 0};
  i32x8 z8 = {0, 0, 0, 0, 0, 0, 0, 0};
  __builtin_amdgcn_tensor_load_to_lds(g0, g1, z4, z4, z8, 0);
}

__global__ void __launch_bounds__(512)
k_qkv_attn(const uint16_t* __restrict__ xp,
           const uint16_t* __restrict__ wqkvT,
           uint16_t* __restrict__ feats) {
  extern __shared__ float smem[];
  float* qs = smem;
  float* ks = smem + LDS_QKV;
  float* vs = smem + 2 * LDS_QKV;
  float* sc = smem + SC_F;                       // [32][3][3]
  uint16_t* aBuf = (uint16_t*)(smem + ABUF_F);   // 2 x [96][40] bf16

  const int tid  = threadIdx.x;
  const int wave = tid >> 5;          // 0..15 -> column tile
  const int lane = tid & 31;
  const int lr   = lane & 15;
  const int lh   = lane >> 4;
  const int bt   = blockIdx.x;        // batch tile
  const int head = blockIdx.y;

  const uint16_t* Abase = xp + (size_t)bt * MROWS * IN_DIM;

  v8f acc[3][MT] = {};                // [qkv][mtile], one 16-col N-tile/wave

  // Preload first A tile via TDM.
  if (wave == 0) {
    tdm_load_a(Abase, ABUF_BYTE);
    __builtin_amdgcn_s_wait_tensorcnt(0);
  }
  __syncthreads();

  for (int k0 = 0; k0 < IN_DIM; k0 += 32) {
    const int cur = (k0 >> 5) & 1;
    if (wave == 0 && k0 + 32 < IN_DIM)
      tdm_load_a(Abase + k0 + 32, ABUF_BYTE + (cur ^ 1) * ABUF_B);

    v16bf b[3];
#pragma unroll
    for (int pi = 0; pi < 3; ++pi) {
      const uint16_t* Bbase =
          wqkvT + (size_t)(((pi * HEADS + head) << 8) + wave * 16) * IN_DIM;
      b[pi] = load_b_frag(Bbase, IN_DIM, lr, lh, k0);
    }

    const uint16_t* ab = aBuf + cur * ABUF_E;
#pragma unroll
    for (int mt = 0; mt < MT; ++mt) {
      const uint16_t* p = ab + (mt * 16 + lr) * A_ROWE + 8 * lh;
      v16bf a = cat16(*reinterpret_cast<const v8bf*>(p),
                      *reinterpret_cast<const v8bf*>(p + 16));
      acc[0][mt] = WMMA_BF16(a, b[0], acc[0][mt]);
      acc[1][mt] = WMMA_BF16(a, b[1], acc[1][mt]);
      acc[2][mt] = WMMA_BF16(a, b[2], acc[2][mt]);
    }

    if (wave == 0 && k0 + 32 < IN_DIM)
      __builtin_amdgcn_s_wait_tensorcnt(0);
    __syncthreads();
  }

  // Spill q/k/v accumulators to LDS (f32).
#pragma unroll
  for (int pi = 0; pi < 3; ++pi) {
    float* dst = (pi == 0) ? qs : (pi == 1) ? ks : vs;
    int col = wave * 16 + lr;
#pragma unroll
    for (int mt = 0; mt < MT; ++mt)
#pragma unroll
      for (int i = 0; i < 8; ++i) {
        int m = mt * 16 + i + 8 * lh;
        dst[m * LDSTR + col] = acc[pi][mt][i];
      }
  }
  __syncthreads();

  // scores[b][i][j] = (q_i . k_j) / 16  (+ mask i>j)
  if (tid < BT * 9) {
    int b = tid / 9, r = tid % 9, i = r / 3, j = r % 3;
    const float* qp = qs + (b * 3 + i) * LDSTR;
    const float* kp = ks + (b * 3 + j) * LDSTR;
    float s = 0.f;
#pragma unroll 4
    for (int c = 0; c < OUT_DIM; c += 4)
      s += qp[c] * kp[c] + qp[c + 1] * kp[c + 1] +
           qp[c + 2] * kp[c + 2] + qp[c + 3] * kp[c + 3];
    s *= 0.0625f;                     // 1/sqrt(256)
    if (i > j) s += NEGINF;
    sc[tid] = s;
  }
  __syncthreads();

  // softmax over i (axis=2 of [B,H,i,j]) per (b, j).
  if (tid < BT * 3) {
    int b = tid / 3, j = tid % 3;
    float s0 = sc[b * 9 + 0 + j];
    float s1 = sc[b * 9 + 3 + j];
    float s2 = sc[b * 9 + 6 + j];
    float m  = fmaxf(s0, fmaxf(s1, s2));
    float e0 = __expf(s0 - m), e1 = __expf(s1 - m), e2 = __expf(s2 - m);
    float inv = 1.0f / (e0 + e1 + e2);
    sc[b * 9 + 0 + j] = e0 * inv;
    sc[b * 9 + 3 + j] = e1 * inv;
    sc[b * 9 + 6 + j] = e2 * inv;
  }
  __syncthreads();

  // z[b][j][o] = sum_i attn[b][i][j] * v[b][i][o]; write feats head-major bf16.
  for (int idx = tid; idx < BT * N_SEQ * OUT_DIM; idx += blockDim.x) {
    int b = idx / (N_SEQ * OUT_DIM);
    int r = idx % (N_SEQ * OUT_DIM);
    int j = r >> 8;
    int o = r & 255;
    float a0 = sc[b * 9 + 0 + j];
    float a1 = sc[b * 9 + 3 + j];
    float a2 = sc[b * 9 + 6 + j];
    const float* vp = vs + (b * 3) * LDSTR + o;
    float z = a0 * vp[0] + a1 * vp[LDSTR] + a2 * vp[2 * LDSTR];
    size_t row = (size_t)(bt * BT + b) * N_SEQ + j;
    feats[row * FEAT + (size_t)head * OUT_DIM + o] = f2bf(z);
  }
}

// ---------------------------------------------------------------------------
// FFN GEMM: out[ROWS][256] = feats[ROWS][2048] @ WfT^T + bf.
// grid = ROWS/128; block = 512 (16 waves).  Wave w: rows [(w/2)*16, +16),
// column half (w&1)*128 -> 8 acc tiles (64 VGPRs): no spilling.
// ---------------------------------------------------------------------------
__global__ void __launch_bounds__(512)
k_ffn(const uint16_t* __restrict__ feats,
      const uint16_t* __restrict__ wfT,
      const float* __restrict__ bias,
      float* __restrict__ out) {
  const int wave = threadIdx.x >> 5;  // 0..15
  const int lane = threadIdx.x & 31;
  const int lr   = lane & 15;
  const int lh   = lane >> 4;
  const int ch   = wave & 1;          // column half: 0 or 1 (128 cols each)
  const size_t r0 = (size_t)blockIdx.x * 128 + (size_t)(wave >> 1) * 16;

  v8f acc[8] = {};
  const uint16_t* Ab = feats + r0 * FEAT;
  const uint16_t* Bb = wfT + (size_t)(ch * 128) * FEAT;

  for (int k0 = 0; k0 < FEAT; k0 += 32) {
    if (k0 + 64 < FEAT)
      __builtin_prefetch(Ab + (size_t)lr * FEAT + k0 + 64, 0, 3);
    v16bf a = load_a_frag(Ab, FEAT, lr, lh, k0);
#pragma unroll
    for (int nt = 0; nt < 8; ++nt) {
      v16bf b = load_b_frag(Bb + (size_t)nt * 16 * FEAT, FEAT, lr, lh, k0);
      acc[nt] = WMMA_BF16(a, b, acc[nt]);
    }
  }

#pragma unroll
  for (int nt = 0; nt < 8; ++nt) {
    int col = ch * 128 + nt * 16 + lr;
    float bv = bias[col];
#pragma unroll
    for (int i = 0; i < 8; ++i) {
      size_t row = r0 + i + 8 * lh;
      out[row * OUT_DIM + col] = acc[nt][i] + bv;
    }
  }
}

// ---------------------------------------------------------------------------
// Host launcher.
// ---------------------------------------------------------------------------
extern "C" void kernel_launch(void* const* d_in, const int* in_sizes, int n_in,
                              void* d_out, int out_size, void* d_ws, size_t ws_size,
                              hipStream_t stream) {
  (void)in_sizes; (void)n_in; (void)out_size; (void)ws_size;
  const float* x   = (const float*)d_in[0];
  const float* pos = (const float*)d_in[1];
  const float* Wq  = (const float*)d_in[2];
  const float* Wk  = (const float*)d_in[3];
  const float* Wv  = (const float*)d_in[4];
  const float* Wf  = (const float*)d_in[5];
  const float* bf  = (const float*)d_in[6];
  float* out = (float*)d_out;

  // Workspace carve-up (256B aligned).
  uint8_t* ws = (uint8_t*)d_ws;
  size_t off = 0;
  auto carve = [&](size_t bytes) -> uint8_t* {
    uint8_t* p = ws + off;
    off = (off + bytes + 255) & ~(size_t)255;
    return p;
  };
  uint16_t* xp    = (uint16_t*)carve((size_t)ROWS * IN_DIM * 2);    // 100.7 MB
  uint16_t* wqkvT = (uint16_t*)carve((size_t)QKV_N * IN_DIM * 2);   //  12.6 MB
  uint16_t* wfT   = (uint16_t*)carve((size_t)OUT_DIM * FEAT * 2);   //   1.0 MB
  uint16_t* feats = (uint16_t*)carve((size_t)ROWS * FEAT * 2);      // 201.3 MB

  // Stage 0: element-wise prep / weight repack to bf16.
  {
    size_t t = (size_t)ROWS * IN_DIM / 8;
    k_prep_xp<<<dim3((unsigned)((t + 255) / 256)), dim3(256), 0, stream>>>(x, pos, xp);
  }
  {
    size_t t = (size_t)QKV_N * IN_DIM;
    k_pack_wqkv<<<dim3((unsigned)((t + 255) / 256)), dim3(256), 0, stream>>>(Wq, Wk, Wv, wqkvT);
  }
  {
    size_t t = (size_t)OUT_DIM * FEAT;
    k_pack_wf<<<dim3((unsigned)((t + 255) / 256)), dim3(256), 0, stream>>>(Wf, wfT);
  }

  // Stage 1: fused QKV projection + attention (bf16 WMMA, TDM-staged A).
  {
    dim3 grid(B_SZ / BT, HEADS);
    k_qkv_attn<<<grid, dim3(512), (size_t)LDS_BYTES, stream>>>(xp, wqkvT, feats);
  }

  // Stage 2: FFN GEMM + bias.
  {
    dim3 grid(ROWS / 128);
    k_ffn<<<grid, dim3(512), 0, stream>>>(feats, wfT, bf, out);
  }
}
